// GraphAttentionLayer_66305705116166
// MI455X (gfx1250) — matrix operand order repaired and checked by reading, not compiled, not run on hardware
//
#include <hip/hip_runtime.h>

// CDNA5 (gfx1250) wave32 WMMA types
typedef __attribute__((ext_vector_type(16))) _Float16 v16h;
typedef __attribute__((ext_vector_type(8)))  float    v8f;

#define ALPHA   0.2f
#define NEG_INF -1000000000000.0f

constexpr int B_   = 4;
constexpr int N_   = 4096;
constexpr int FIN  = 256;
constexpr int FOUT = 128;

// -----------------------------------------------------------------------------
// Kernel 1: h = inp @ W via v_wmma_f32_16x16x32_f16.
// Output stored TRANSPOSED as f16: hT[b][f][n]  (f = 0..127, n = 0..4095)
// so the attention kernel's WMMA B-operand is a contiguous 32B load per lane.
//
// WMMA A operand as loaded here: lane<16 -> row=lane, halves 0..15 = K
// k0..k0+15 (16 consecutive); lane>=16 -> row=lane-16, halves 0..15 = K
// k0+16..k0+31 (contiguous K, not the ISA table order).  B, same K order: lane -> col=lane&15, lane<16 holds K
// k0..k0+15 (16 consecutive), lane>=16 holds K k0+16..k0+31.
// -----------------------------------------------------------------------------
__global__ __launch_bounds__(256) void gat_h_kernel(const float* __restrict__ inp,
                                                    const float* __restrict__ W,
                                                    _Float16* __restrict__ hT)
{
    const int tid  = threadIdx.x;
    const int wave = tid >> 5;
    const int lane = tid & 31;
    const int r15  = lane & 15;
    const int hi   = lane >> 4;          // 0 or 1

    const int rowBase = blockIdx.x * 16; // over B*N = 16384 rows (16 | 4096)
    const int colBase = wave * 16;       // 8 waves cover all 128 output cols
    const int col     = colBase + r15;

    v8f acc = {};
    const float* arow = inp + (size_t)(rowBase + r15) * FIN;

    for (int k0 = 0; k0 < FIN; k0 += 32) {
        // ---- A operand: inp rows, f32 -> f16 ----
        const int base1 = k0 + hi * 16;
        float4 f0 = *(const float4*)(arow + base1);
        float4 f1 = *(const float4*)(arow + base1 + 4);
        float4 f2 = *(const float4*)(arow + base1 + 8);
        float4 f3 = *(const float4*)(arow + base1 + 12);
        v16h va;
        va[0]=(_Float16)f0.x;  va[1]=(_Float16)f0.y;  va[2]=(_Float16)f0.z;  va[3]=(_Float16)f0.w;
        va[4]=(_Float16)f1.x;  va[5]=(_Float16)f1.y;  va[6]=(_Float16)f1.z;  va[7]=(_Float16)f1.w;
        va[8]=(_Float16)f2.x;  va[9]=(_Float16)f2.y;  va[10]=(_Float16)f2.z; va[11]=(_Float16)f2.w;
        va[12]=(_Float16)f3.x; va[13]=(_Float16)f3.y; va[14]=(_Float16)f3.z; va[15]=(_Float16)f3.w;

        // ---- B operand: W column 'col', 16 consecutive K (strided, L2-hot) ----
        const int kk = k0 + hi * 16;
        v16h vb;
        #pragma unroll
        for (int i = 0; i < 16; ++i)
            vb[i] = (_Float16)W[(size_t)(kk + i) * FOUT + col];

        acc = __builtin_amdgcn_wmma_f32_16x16x32_f16(
                  false, va, false, vb, (short)0, acc, false, false);
    }

    // C/D layout: VGPR r -> row M = r + 8*hi, col = lane&15
    #pragma unroll
    for (int r = 0; r < 8; ++r) {
        const int row = rowBase + r + hi * 8;
        const int bb  = row >> 12;
        const int n   = row & (N_ - 1);
        hT[(((size_t)(bb * FOUT + col)) << 12) + n] = (_Float16)acc[r];
    }
}

// -----------------------------------------------------------------------------
// Kernel 2: s1[b,n] = sum_f h[b,n,f]*a1[f], s2 likewise with a2.
// Reads hT (transposed): lane-consecutive n -> coalesced per f.
// -----------------------------------------------------------------------------
__global__ __launch_bounds__(256) void gat_s_kernel(const _Float16* __restrict__ hT,
                                                    const float* __restrict__ a,
                                                    float* __restrict__ s1,
                                                    float* __restrict__ s2)
{
    const int row = blockIdx.x * 256 + threadIdx.x;  // 0..16383
    const int bb  = row >> 12;
    const int n   = row & (N_ - 1);
    const _Float16* base = hT + (((size_t)bb * FOUT) << 12) + n;

    float sum1 = 0.f, sum2 = 0.f;
    #pragma unroll 4
    for (int f = 0; f < FOUT; ++f) {
        float v = (float)base[(size_t)f << 12];
        sum1 += v * a[f];
        sum2 += v * a[FOUT + f];
    }
    s1[row] = sum1;
    s2[row] = sum2;
}

// -----------------------------------------------------------------------------
// Kernel 3: fused attention (flash-style online softmax + WMMA P@h).
// One block = 256 threads = 8 waves handles a 16-row tile of one batch.
// Wave w accumulates output columns [16w, 16w+16).  adj streamed exactly once.
// -----------------------------------------------------------------------------
__global__ __launch_bounds__(256) void gat_attn_kernel(const float* __restrict__ adj,
                                                       const _Float16* __restrict__ hT,
                                                       const float* __restrict__ s1,
                                                       const float* __restrict__ s2,
                                                       float* __restrict__ out)
{
    __shared__ float pmax[16][16];                 // per-row partial maxima
    __shared__ alignas(32) _Float16 aBuf[32 * 16]; // P tile in A-operand layout
    __shared__ float sm[16], sscale[16], sl[16], ssum[16], ss1[16];

    const int tid  = threadIdx.x;
    const int wave = tid >> 5;
    const int lane = tid & 31;
    const int r15  = lane & 15;
    const int hi   = lane >> 4;

    const int blk = blockIdx.x;
    const int bb  = blk >> 8;            // batch
    const int i0  = (blk & 255) * 16;    // row tile base within batch
    const int colBase = wave * 16;

    const float* adjB   = adj + ((size_t)bb << 24);          // b*N*N
    const float* s2B    = s2 + bb * N_;
    const _Float16* hTB = hT + (((size_t)bb * FOUT) << 12);

    if (tid < 16) {
        ss1[tid]  = s1[bb * N_ + i0 + tid];
        sm[tid]   = -__builtin_inff();
        sl[tid]   = 0.0f;
        ssum[tid] = 0.0f;   // deferred (unscaled) row-sum of current step
    }
    v8f acc = {};
    __syncthreads();

    const int r  = tid >> 4;        // 0..15 : attention row
    const int c0 = (tid & 15) * 2;  // 0..30 : even column within 32-wide tile

    for (int j0 = 0; j0 < N_; j0 += 32) {
        // ---- step 1: raw attention values for this 16x32 tile (registers) ----
        const float* arow = adjB + (size_t)(i0 + r) * N_ + j0 + c0;
        float2 av = *(const float2*)arow;
        if (j0 + 32 < N_) __builtin_prefetch(arow + 32, 0, 1); // global_prefetch_b8
        const float s1r = ss1[r];
        float e0 = s1r + s2B[j0 + c0];
        float e1 = s1r + s2B[j0 + c0 + 1];
        e0 = e0 > 0.f ? e0 : ALPHA * e0;
        e1 = e1 > 0.f ? e1 : ALPHA * e1;
        const float t0 = av.x > 0.f ? av.x * e0 : NEG_INF;
        const float t1 = av.y > 0.f ? av.y * e1 : NEG_INF;
        pmax[r][tid & 15] = fmaxf(t0, t1);
        __syncthreads();

        // ---- step 2: online max / rescale bookkeeping (rows 0..15) ----
        if (tid < 16) {
            const float mold = sm[tid];
            float tmax = pmax[tid][0];
            #pragma unroll
            for (int k = 1; k < 16; ++k) tmax = fmaxf(tmax, pmax[tid][k]);
            const float mnew = fmaxf(mold, tmax);
            const float sc   = __expf(mold - mnew);      // 0 on first step
            sl[tid]     = (sl[tid] + ssum[tid]) * sc;    // fold in prev step's sum
            ssum[tid]   = 0.0f;
            sscale[tid] = sc;
            sm[tid]     = mnew;
        }
        __syncthreads();

        // ---- step 3: P = exp(att - m) -> f16 into WMMA A-operand layout ----
        {
            const float mr = sm[r];
            const float p0 = __expf(t0 - mr);
            const float p1 = __expf(t1 - mr);
            atomicAdd(&ssum[r], p0 + p1);                // ds_add_f32
            // (r,c) -> lane = r + 16*bit4(c), half = c&15
            const int lane0 = r + ((c0 & 16) ? 16 : 0);
            const int half0 = c0 & 15;
            aBuf[lane0 * 16 + half0]     = (_Float16)p0;
            aBuf[lane0 * 16 + half0 + 1] = (_Float16)p1;
        }
        // rescale accumulators by exp(m_old - m_new) per output row
        {
            const int rofs = hi << 3;
            #pragma unroll
            for (int rr = 0; rr < 8; ++rr) acc[rr] *= sscale[rr + rofs];
        }
        __syncthreads();

        // ---- step 4: acc += P(16x32) @ h(32x16) per wave ----
        v16h pa = *(const v16h*)(aBuf + lane * 16);                       // LDS, 32B/lane
        const _Float16* bp = hTB + (((size_t)(colBase + r15)) << 12) + j0 + hi * 16;
        v16h pb = *(const v16h*)bp;                                       // contiguous 32B
        acc = __builtin_amdgcn_wmma_f32_16x16x32_f16(
                  false, pa, false, pb, (short)0, acc, false, false);
        __syncthreads();
    }

    // ---- finalize: /l then ELU ----
    const int rofs = hi << 3;
    #pragma unroll
    for (int rr = 0; rr < 8; ++rr) {
        const int M = rr + rofs;
        const float linv = 1.0f / (sl[M] + ssum[M]);
        float v = acc[rr] * linv;
        v = v > 0.f ? v : __expf(v) - 1.0f;
        out[((size_t)(bb * N_ + i0 + M)) * FOUT + colBase + r15] = v;
    }
}

// -----------------------------------------------------------------------------
// Launcher
// -----------------------------------------------------------------------------
extern "C" void kernel_launch(void* const* d_in, const int* in_sizes, int n_in,
                              void* d_out, int out_size, void* d_ws, size_t ws_size,
                              hipStream_t stream)
{
    const float* inp = (const float*)d_in[0];   // (4, 4096, 256)
    const float* adj = (const float*)d_in[1];   // (4, 4096, 4096)
    const float* W   = (const float*)d_in[2];   // (256, 128)
    const float* a   = (const float*)d_in[3];   // (256, 1)
    float* out = (float*)d_out;                 // (4, 4096, 128)

    char* ws = (char*)d_ws;
    _Float16* hT = (_Float16*)ws;                              // 4 MB (b,f,n) f16
    float* s1    = (float*)(ws + (size_t)4 * 1024 * 1024);     // 64 KB
    float* s2    = s1 + (size_t)B_ * N_;                       // 64 KB

    gat_h_kernel   <<<dim3((B_ * N_) / 16), dim3(256), 0, stream>>>(inp, W, hT);
    gat_s_kernel   <<<dim3((B_ * N_) / 256), dim3(256), 0, stream>>>(hT, a, s1, s2);
    gat_attn_kernel<<<dim3(B_ * (N_ / 16)), dim3(256), 0, stream>>>(adj, hT, s1, s2, out);
}
